// CrossAttention_12240656794270
// MI455X (gfx1250) — compile-verified, matrix operand order checked by reading, not run on hardware
//
#include <hip/hip_runtime.h>

typedef __attribute__((ext_vector_type(16))) _Float16 v16h;
typedef __attribute__((ext_vector_type(8)))  _Float16 v8h;
typedef __attribute__((ext_vector_type(8)))  float    v8f;
typedef __attribute__((ext_vector_type(4)))  float    v4f;
typedef __attribute__((ext_vector_type(4)))  int      v4i;

#define DIM     512
#define NHEADS  8
#define HD      64
#define BATCH   4
#define SEQ     2048

// ---------------- CDNA5 feature detection (compile-safe fallbacks) ----------
#if defined(__HIP_DEVICE_COMPILE__) && \
    __has_builtin(__builtin_amdgcn_global_load_async_to_lds_b128) && \
    __has_builtin(__builtin_amdgcn_s_wait_asynccnt)
#define USE_ASYNC_LDS 1
#else
#define USE_ASYNC_LDS 0
#endif

#if defined(__HIP_DEVICE_COMPILE__) && \
    __has_builtin(__builtin_amdgcn_mov_dpp8) && \
    __has_builtin(__builtin_amdgcn_update_dpp)
#define USE_DPP 1
#else
#define USE_DPP 0
#endif

#if USE_ASYNC_LDS
// Builtin signature (from compiler diagnostic): param0 = addrspace(1) int4*
// (non-const), so: global src via value-preserving integer round-trip,
// LDS dst via proper addrspacecast.
typedef __attribute__((address_space(1))) v4i gv4i;
typedef __attribute__((address_space(3))) v4i lv4i;
#define ASYNC_B128(gp, lp)                                      \
  __builtin_amdgcn_global_load_async_to_lds_b128(               \
      (gv4i*)(unsigned long long)(gp), (lv4i*)(lp), 0, 0)
#endif

static __device__ __forceinline__ v16h cat8(v8h lo, v8h hi) {
  return __builtin_shufflevector(lo, hi, 0,1,2,3,4,5,6,7,8,9,10,11,12,13,14,15);
}

static __device__ __forceinline__ v8f wmma_f16(v16h a, v16h b, v8f c) {
  // D = A(16x32 f16) x B(32x16 f16) + C(16x16 f32)
  return __builtin_amdgcn_wmma_f32_16x16x32_f16(
      /*neg_a=*/false, a, /*neg_b=*/false, b,
      /*c_mod=*/(short)0, c, /*reuse_a=*/false, /*reuse_b=*/false);
}

// ---------------- 16-lane row reductions (pure VALU via DPP on wave32) ------
#if USE_DPP
constexpr int DPP8SEL(int a,int b,int c,int d,int e,int f,int g,int h) {
  return a | (b<<3) | (c<<6) | (d<<9) | (e<<12) | (f<<15) | (g<<18) | (h<<21);
}
#define DPP_XOR1 DPP8SEL(1,0,3,2,5,4,7,6)
#define DPP_XOR2 DPP8SEL(2,3,0,1,6,7,4,5)
#define DPP_XOR4 DPP8SEL(4,5,6,7,0,1,2,3)

template <int SEL>
static __device__ __forceinline__ float fdpp8(float x) {
  return __int_as_float(__builtin_amdgcn_mov_dpp8(__float_as_int(x), SEL));
}
static __device__ __forceinline__ float frow_ror8(float x) {
  const int i = __float_as_int(x);
  // DPP16 ROW_ROR:8 (0x128): rotate within the 16-lane row -> swaps 8-groups
  return __int_as_float(__builtin_amdgcn_update_dpp(i, i, 0x128, 0xf, 0xf, false));
}
#endif

static __device__ __forceinline__ float rowred_max16(float x) {
#if USE_DPP
  x = fmaxf(x, fdpp8<DPP_XOR1>(x));
  x = fmaxf(x, fdpp8<DPP_XOR2>(x));
  x = fmaxf(x, fdpp8<DPP_XOR4>(x));
  x = fmaxf(x, frow_ror8(x));
#else
  #pragma unroll
  for (int off = 1; off < 16; off <<= 1) x = fmaxf(x, __shfl_xor(x, off, 32));
#endif
  return x;
}
static __device__ __forceinline__ float rowred_sum16(float x) {
#if USE_DPP
  x += fdpp8<DPP_XOR1>(x);
  x += fdpp8<DPP_XOR2>(x);
  x += fdpp8<DPP_XOR4>(x);
  x += frow_ror8(x);
#else
  #pragma unroll
  for (int off = 1; off < 16; off <<= 1) x += __shfl_xor(x, off, 32);
#endif
  return x;
}

// ---------------------------------------------------------------------------
// GEMM: Out[M x 512] = A[M x 512] @ W[512 x 512]^T + bias
// Block tile 128(M) x 64(N), K-step 32, double-buffered LDS (1 barrier/step).
// 256 threads = 8 waves; each wave owns a 16x64 strip (4 WMMA accumulators).
// ---------------------------------------------------------------------------
template <bool A_IS_F32, bool OUT_IS_F32>
__global__ __launch_bounds__(256) void gemm_bias_kernel(
    const void* __restrict__ Aptr, const float* __restrict__ W,
    const float* __restrict__ bias, void* __restrict__ Out, int M) {
  __shared__ _Float16 As[2][128 * 40];  // [m][k], 40-half stride (16B rows)
  __shared__ _Float16 Ws[2][64 * 40];   // [n][k]

  const int t    = threadIdx.x;
  const int lane = t & 31;
  const int wv   = t >> 5;  // 0..7
  const int m0   = blockIdx.x * 128;
  const int n0   = blockIdx.y * 64;

  v8f acc[4] = {};

  for (int k0 = 0, buf = 0; k0 < DIM; k0 += 32, buf ^= 1) {
    // --- stage A tile 128x32 ---
    if constexpr (A_IS_F32) {
      const float* A = (const float*)Aptr;
      #pragma unroll
      for (int p = 0; p < 4; ++p) {
        int i = p * 256 + t;           // 1024 float4 chunks (8 per row)
        int r = i >> 3, c4 = i & 7;
        v4f v = *(const v4f*)(A + (size_t)(m0 + r) * DIM + k0 + c4 * 4);
        _Float16* dst = &As[buf][r * 40 + c4 * 4];
        dst[0] = (_Float16)v.x; dst[1] = (_Float16)v.y;
        dst[2] = (_Float16)v.z; dst[3] = (_Float16)v.w;
      }
    } else {
      const _Float16* A = (const _Float16*)Aptr;
      #pragma unroll
      for (int p = 0; p < 2; ++p) {
        int i = p * 256 + t;           // 512 v8h chunks (4 per row)
        int r = i >> 2, c8 = i & 3;
        const _Float16* src = A + (size_t)(m0 + r) * DIM + k0 + c8 * 8;
#if USE_ASYNC_LDS
        ASYNC_B128(src, &As[buf][r * 40 + c8 * 8]);
#else
        *(v8h*)&As[buf][r * 40 + c8 * 8] = *(const v8h*)src;
#endif
      }
    }
    // --- stage W tile 64x32 (f32 -> f16) ---
    #pragma unroll
    for (int p = 0; p < 2; ++p) {
      int i = p * 256 + t;             // 512 float4 chunks (8 per row)
      int r = i >> 3, c4 = i & 7;
      v4f v = *(const v4f*)(W + (size_t)(n0 + r) * DIM + k0 + c4 * 4);
      _Float16* dst = &Ws[buf][r * 40 + c4 * 4];
      dst[0] = (_Float16)v.x; dst[1] = (_Float16)v.y;
      dst[2] = (_Float16)v.z; dst[3] = (_Float16)v.w;
    }
    // --- prefetch next K-tile HBM lines ---
    if (k0 + 32 < DIM) {
      if (t < 128) {
        if constexpr (A_IS_F32)
          __builtin_prefetch((const float*)Aptr + (size_t)(m0 + t) * DIM + k0 + 32, 0, 1);
        else
          __builtin_prefetch((const _Float16*)Aptr + (size_t)(m0 + t) * DIM + k0 + 32, 0, 1);
      } else if (t < 192) {
        __builtin_prefetch(W + (size_t)(n0 + t - 128) * DIM + k0 + 32, 0, 1);
      }
    }
#if USE_ASYNC_LDS
    if constexpr (!A_IS_F32) __builtin_amdgcn_s_wait_asynccnt(0);
#endif
    __syncthreads();

    // --- WMMA: A fragment (16x32), 4 B fragments (32x16) ---
    const int am = wv * 16 + (lane & 15);
    const int kb = 8 * (lane >> 4);    // lo chunk k-base; hi chunk at +16
    v16h a = cat8(*(const v8h*)&As[buf][am * 40 + kb],
                  *(const v8h*)&As[buf][am * 40 + 16 + kb]);
    #pragma unroll
    for (int nt = 0; nt < 4; ++nt) {
      const int bn = nt * 16 + (lane & 15);
      v16h b = cat8(*(const v8h*)&Ws[buf][bn * 40 + kb],
                    *(const v8h*)&Ws[buf][bn * 40 + 16 + kb]);
      acc[nt] = wmma_f16(a, b, acc[nt]);
    }
  }

  // --- epilogue. C/D layout: lane l, vgpr r -> M = r + 8*(l>>4), N = l&15
  #pragma unroll
  for (int nt = 0; nt < 4; ++nt) {
    const int col = n0 + nt * 16 + (lane & 15);
    const float bv = bias[col];
    #pragma unroll
    for (int r = 0; r < 8; ++r) {
      const int row = m0 + wv * 16 + r + 8 * (lane >> 4);
      const float val = acc[nt][r] + bv;
      if constexpr (OUT_IS_F32)
        ((float*)Out)[(size_t)row * DIM + col] = val;
      else
        ((_Float16*)Out)[(size_t)row * DIM + col] = (_Float16)val;
    }
  }
}

// ---------------------------------------------------------------------------
// Flash attention: one block per (b, h, 64 q-rows). 128 threads = 4 waves,
// each wave owns 16 q-rows. KV chunks of 32, double-buffered (1 barrier/chunk),
// online softmax with DPP row reductions.
// ---------------------------------------------------------------------------
__global__ __launch_bounds__(128) void flash_attn_kernel(
    const _Float16* __restrict__ Qh, const _Float16* __restrict__ Kh,
    const _Float16* __restrict__ Vh, _Float16* __restrict__ Oh) {
  __shared__ _Float16 Qs[64 * 72];         // [q][d]
  __shared__ _Float16 Ks[2][32 * 72];      // [kv][d]  (k-dim of QK^T contiguous)
  __shared__ _Float16 Vt[2][64 * 40];      // [d][kv]  (k-dim of PV contiguous)
  __shared__ _Float16 Ps[4][16 * 40];      // per-wave P tile [m][kv]

  const int t    = threadIdx.x;
  const int lane = t & 31;
  const int wv   = t >> 5;              // 0..3
  const int q0   = blockIdx.x * 64;
  const int bh   = blockIdx.y;
  const int b    = bh >> 3;
  const int h    = bh & 7;
  const size_t qbase  = ((size_t)b * SEQ + q0) * DIM + (size_t)h * HD;
  const size_t kvbase = ((size_t)b * SEQ) * DIM + (size_t)h * HD;

  // stage Q tile 64x64 (512 v8h chunks, 8 per row)
  #pragma unroll
  for (int p = 0; p < 4; ++p) {
    int i = p * 128 + t;
    int r = i >> 3, c8 = i & 7;
    const _Float16* src = Qh + qbase + (size_t)r * DIM + c8 * 8;
#if USE_ASYNC_LDS
    ASYNC_B128(src, &Qs[r * 72 + c8 * 8]);
#else
    *(v8h*)&Qs[r * 72 + c8 * 8] = *(const v8h*)src;
#endif
  }

  float rmax[8], rsum[8];
  v8f o[4] = {};
  #pragma unroll
  for (int r = 0; r < 8; ++r) { rmax[r] = -3.0e38f; rsum[r] = 0.0f; }
  const float cexp = 0.125f * 1.44269504088896f;  // SCALE * log2(e)

  const int kb = 8 * (lane >> 4);
  const int nl = lane & 15;
  const int am = wv * 16 + nl;

  for (int kv0 = 0, buf = 0; kv0 < SEQ; kv0 += 32, buf ^= 1) {
    // stage K chunk [kv][d] (raw copy -> async) and V chunk transposed [d][kv]
    #pragma unroll
    for (int p = 0; p < 2; ++p) {
      int i = p * 128 + t;
      int r = i >> 3, c8 = i & 7;
      const _Float16* ksrc = Kh + kvbase + (size_t)(kv0 + r) * DIM + c8 * 8;
#if USE_ASYNC_LDS
      ASYNC_B128(ksrc, &Ks[buf][r * 72 + c8 * 8]);
#else
      *(v8h*)&Ks[buf][r * 72 + c8 * 8] = *(const v8h*)ksrc;
#endif
      v8h vvv = *(const v8h*)(Vh + kvbase + (size_t)(kv0 + r) * DIM + c8 * 8);
      #pragma unroll
      for (int j = 0; j < 8; ++j) Vt[buf][(c8 * 8 + j) * 40 + r] = vvv[j];
    }
    // prefetch next chunk's K/V HBM lines (one 128B row per thread)
    if (kv0 + 32 < SEQ && t < 64) {
      const _Float16* bsrc = (t < 32) ? Kh : Vh;
      __builtin_prefetch(bsrc + kvbase + (size_t)(kv0 + 32 + (t & 31)) * DIM, 0, 1);
    }
#if USE_ASYNC_LDS
    __builtin_amdgcn_s_wait_asynccnt(0);
#endif
    __syncthreads();

    // S(16x32) = Q_tile @ K_chunk^T  (two k-steps over d, two n-tiles over kv)
    v8f s0 = {}, s1 = {};
    #pragma unroll
    for (int ks = 0; ks < 2; ++ks) {
      const int d0 = ks * 32;
      v16h a  = cat8(*(const v8h*)&Qs[am * 72 + d0 + kb],
                     *(const v8h*)&Qs[am * 72 + d0 + 16 + kb]);
      v16h b0 = cat8(*(const v8h*)&Ks[buf][nl * 72 + d0 + kb],
                     *(const v8h*)&Ks[buf][nl * 72 + d0 + 16 + kb]);
      v16h b1 = cat8(*(const v8h*)&Ks[buf][(16 + nl) * 72 + d0 + kb],
                     *(const v8h*)&Ks[buf][(16 + nl) * 72 + d0 + 16 + kb]);
      s0 = wmma_f16(a, b0, s0);
      s1 = wmma_f16(a, b1, s1);
    }

    // online softmax. Row of vgpr r lives in one 16-lane half (M = r+8*(l>>4));
    // DPP xor1/2/4 + ROW_ROR:8 reductions never cross the half boundary.
    float p0[8], p1[8], alpha[8];
    #pragma unroll
    for (int r = 0; r < 8; ++r) {
      const float mx   = rowred_max16(fmaxf(s0[r], s1[r]));
      const float mnew = fmaxf(rmax[r], mx);
      const float al   = exp2f((rmax[r] - mnew) * cexp);
      p0[r] = exp2f((s0[r] - mnew) * cexp);
      p1[r] = exp2f((s1[r] - mnew) * cexp);
      rsum[r]  = rsum[r] * al + rowred_sum16(p0[r] + p1[r]);
      rmax[r]  = mnew;
      alpha[r] = al;
    }
    #pragma unroll
    for (int dt = 0; dt < 4; ++dt)
      #pragma unroll
      for (int r = 0; r < 8; ++r) o[dt][r] *= alpha[r];

    // bounce P through LDS: C-layout -> A-layout. Wave-private region; same-wave
    // LDS ops are in-order on CDNA5, so no block barrier is needed here.
    #pragma unroll
    for (int r = 0; r < 8; ++r) {
      const int m = r + 8 * (lane >> 4);
      Ps[wv][m * 40 + nl]      = (_Float16)p0[r];
      Ps[wv][m * 40 + 16 + nl] = (_Float16)p1[r];
    }

    // O(16x64) += P(16x32) @ V_chunk(32x64)
    v16h pa = cat8(*(const v8h*)&Ps[wv][nl * 40 + kb],
                   *(const v8h*)&Ps[wv][nl * 40 + 16 + kb]);
    #pragma unroll
    for (int dt = 0; dt < 4; ++dt) {
      const int d = dt * 16 + nl;
      v16h bv = cat8(*(const v8h*)&Vt[buf][d * 40 + kb],
                     *(const v8h*)&Vt[buf][d * 40 + 16 + kb]);
      o[dt] = wmma_f16(pa, bv, o[dt]);
    }
  }

  // normalize and store f16 attention output [B, Sq, DIM]
  #pragma unroll
  for (int r = 0; r < 8; ++r) rsum[r] = 1.0f / rsum[r];
  #pragma unroll
  for (int dt = 0; dt < 4; ++dt) {
    const int col = h * HD + dt * 16 + nl;
    #pragma unroll
    for (int r = 0; r < 8; ++r) {
      const int row = q0 + wv * 16 + r + 8 * (lane >> 4);
      Oh[((size_t)b * SEQ + row) * DIM + col] = (_Float16)(o[dt][r] * rsum[r]);
    }
  }
}

// ---------------------------------------------------------------------------
extern "C" void kernel_launch(void* const* d_in, const int* in_sizes, int n_in,
                              void* d_out, int out_size, void* d_ws, size_t ws_size,
                              hipStream_t stream) {
  (void)in_sizes; (void)n_in; (void)out_size; (void)ws_size;
  const float* x1 = (const float*)d_in[0];
  const float* x2 = (const float*)d_in[1];
  const float* Wq = (const float*)d_in[2];
  const float* bq = (const float*)d_in[3];
  const float* Wk = (const float*)d_in[4];
  const float* bk = (const float*)d_in[5];
  const float* Wv = (const float*)d_in[6];
  const float* bv = (const float*)d_in[7];
  const float* Wo = (const float*)d_in[8];
  const float* bo = (const float*)d_in[9];
  float* out = (float*)d_out;

  const size_t MTOT = (size_t)BATCH * SEQ;        // 8192 rows
  _Float16* Qh = (_Float16*)d_ws;                 //  8 MB
  _Float16* Kh = Qh + MTOT * DIM;                 //  8 MB
  _Float16* Vh = Kh + MTOT * DIM;                 //  8 MB
  _Float16* Ah = Vh + MTOT * DIM;                 //  8 MB

  const int M = (int)MTOT;
  dim3 gg(M / 128, DIM / 64);

  gemm_bias_kernel<true, false><<<gg, 256, 0, stream>>>((const void*)x1, Wq, bq, (void*)Qh, M);
  gemm_bias_kernel<true, false><<<gg, 256, 0, stream>>>((const void*)x2, Wk, bk, (void*)Kh, M);
  gemm_bias_kernel<true, false><<<gg, 256, 0, stream>>>((const void*)x2, Wv, bv, (void*)Vh, M);

  flash_attn_kernel<<<dim3(SEQ / 64, BATCH * NHEADS), 128, 0, stream>>>(Qh, Kh, Vh, Ah);

  gemm_bias_kernel<false, true><<<gg, 256, 0, stream>>>((const void*)Ah, Wo, bo, (void*)out, M);
}
